// RNNModel_19971597926405
// MI455X (gfx1250) — matrix operand order repaired and checked by reading, not compiled, not run on hardware
//
#include <hip/hip_runtime.h>
#include <hip/hip_bf16.h>
#include <math.h>

#define S_LEN 256
#define BATCH 16
#define VOCAB 32000
#define EMB   512
#define HID   512

// LDS row padding to spread bank accesses (row stride 520 bf16 = 1040 B)
#define H_STRIDE    520
#define SLOT_BYTES  (16 * H_STRIDE * 2)
#define STAGE_BYTES (2 * BATCH * HID * 2)   // xc+xh bf16 for one step = 32 KB

typedef __attribute__((ext_vector_type(16))) __bf16 v16bf;
typedef __attribute__((ext_vector_type(8)))  __bf16 v8bf;
typedef __attribute__((ext_vector_type(8)))  float  v8f;

union FragA { v16bf v; v8bf h[2]; __bf16 e[16]; };

__device__ __forceinline__ v8f v8f_zero() {
  v8f z = {0.f, 0.f, 0.f, 0.f, 0.f, 0.f, 0.f, 0.f};
  return z;
}

__device__ __forceinline__ v8f wmma_bf16(v16bf a, v16bf b, v8f c) {
  // (neg_a, A, neg_b, B, c_mod, C, reuse_a, reuse_b)
  return __builtin_amdgcn_wmma_f32_16x16x32_bf16(false, a, false, b, (short)0, c,
                                                 false, false);
}

__device__ __forceinline__ float sigmoidf_(float x) {
  return 1.0f / (1.0f + __expf(-x));
}

__device__ __forceinline__ float actf_(int a, float x) {
  switch (a) {
    case 0:  return tanhf(x);
    case 1:  return fmaxf(x, 0.0f);
    case 2:  return sigmoidf_(x);
    default: return x;  // identity
  }
}

// Async global->LDS stage of one 32 KB pre-activation block (xc|xh, bf16).
// 1024 threads x 2 chunks x 16 B. Tracked on ASYNCcnt, retired by wait_async0.
__device__ __forceinline__ void async_stage(const char* gsrc, char* ldst) {
#pragma unroll
  for (int it = 0; it < 2; ++it) {
    const int off = (int)threadIdx.x * 16 + it * 16384;
    unsigned lds_addr = (unsigned)(size_t)(ldst + off);        // LDS offset = low 32 bits
    unsigned long long ga = (unsigned long long)(size_t)(gsrc + off);
    asm volatile("global_load_async_to_lds_b128 %0, %1, off"
                 :: "v"(lds_addr), "v"(ga) : "memory");
  }
}

__device__ __forceinline__ void wait_async0() {
  asm volatile("s_wait_asynccnt 0x0" ::: "memory");
}

// ---------------------------------------------------------------------------
// Kernel 0: fp32 -> bf16 conversion (weights)
// ---------------------------------------------------------------------------
__global__ void f32_to_bf16(const float* __restrict__ src,
                            __bf16* __restrict__ dst, int n) {
  int i = blockIdx.x * blockDim.x + threadIdx.x;
  int stride = gridDim.x * blockDim.x;
  for (; i < n; i += stride) dst[i] = (__bf16)src[i];
}

// ---------------------------------------------------------------------------
// Kernel 1: embedding gather + x-projections (parallel over S, no recurrence)
//   xpre[t][which][b][j] = x_t @ w.T + bias   (bf16, staged per-step later)
// One wave (32 threads) per (t, 16-col tile, which-matrix). M = B = 16.
// ---------------------------------------------------------------------------
__global__ void embed_xproj(const int* __restrict__ ids,
                            const float* __restrict__ emb,
                            const float* __restrict__ w_xc,
                            const float* __restrict__ b_xc,
                            const float* __restrict__ w_xh,
                            const float* __restrict__ b_xh,
                            __bf16* __restrict__ xpre) {
  const int t = blockIdx.x, jt = blockIdx.y, which = blockIdx.z;
  const float* w    = which ? w_xh : w_xc;
  const float* bias = which ? b_xh : b_xc;
  __bf16* dst = xpre + ((size_t)t * 2 + which) * BATCH * HID;

  const int lane = threadIdx.x;
  const int n = lane & 15, hi = lane >> 4;

  const int id = ids[t * BATCH + n];            // A row r == n (M = lane%16)
  const float* xrow = emb + (size_t)id * EMB;
  const int j = (jt << 4) + n;                  // B column
  const float* wrow = w + (size_t)j * EMB + 16 * hi;

  v8f acc = v8f_zero();
  for (int kk = 0; kk < 16; ++kk) {
    const int k0 = kk * 32;
    FragA a, b;
    const float* p0 = xrow + k0 + 16 * hi;
    const float* p1 = p0 + 8;
#pragma unroll
    for (int i = 0; i < 8; ++i) { a.e[i] = (__bf16)p0[i]; a.e[8 + i] = (__bf16)p1[i]; }
    const float* q = wrow + k0;
#pragma unroll
    for (int i = 0; i < 16; ++i) b.e[i] = (__bf16)q[i];
    acc = wmma_bf16(a.v, b.v, acc);
  }
  const float bj = bias[j];
#pragma unroll
  for (int v = 0; v < 8; ++v) {
    const int m = v + 8 * hi;                   // D[M][N]: M = v + 8*hi
    dst[(size_t)m * HID + j] = (__bf16)(acc[v] + bj);
  }
}

// ---------------------------------------------------------------------------
// Kernel 2: sequential DAG recurrence. One workgroup, 32 waves (wave32),
// DAG node activations live in LDS as bf16 16x520 tiles (5 slots, ~83 KB).
// Per-step pre-activations double-buffered into LDS via async copies.
// Each wave owns one 16-column N-tile of every [16x512]@[512x512] GEMM.
// ---------------------------------------------------------------------------
__device__ void dag_node(const __bf16* hp, __bf16* hd,
                         const __bf16* __restrict__ wc,
                         const __bf16* __restrict__ wh, int act,
                         int lane, int wave) {
  const int n = lane & 15, hi = lane >> 4;
  const int n0 = wave << 4;
  v8f accC = v8f_zero(), accH = v8f_zero();
  const __bf16* arow = hp + n * H_STRIDE;
  const __bf16* wcp = wc + (size_t)(n0 + n) * HID + 16 * hi;
  const __bf16* whp = wh + (size_t)(n0 + n) * HID + 16 * hi;
#pragma unroll 4
  for (int kk = 0; kk < 16; ++kk) {
    const int k0 = kk * 32;
    FragA a;
    a.h[0] = *(const v8bf*)(arow + k0 + 16 * hi);
    a.h[1] = *(const v8bf*)(arow + k0 + 16 * hi + 8);
    v16bf bc = *(const v16bf*)(wcp + k0);
    v16bf bh = *(const v16bf*)(whp + k0);
    accC = wmma_bf16(a.v, bc, accC);
    accH = wmma_bf16(a.v, bh, accH);
  }
#pragma unroll
  for (int v = 0; v < 8; ++v) {
    const int m = v + 8 * hi, col = n0 + n;
    float hpv = (float)hp[m * H_STRIDE + col];
    float c = sigmoidf_(accC[v]);
    float g = actf_(act, accH[v]);
    hd[m * H_STRIDE + col] = (__bf16)(c * g + (1.0f - c) * hpv);
  }
  __syncthreads();
}

__device__ void node0(const __bf16* hp, __bf16* hd,
                      const __bf16* __restrict__ whc,
                      const __bf16* __restrict__ whh,
                      const __bf16* xc, const __bf16* xh,   // LDS staged
                      int lane, int wave) {
  const int n = lane & 15, hi = lane >> 4;
  const int n0 = wave << 4;
  v8f accC = v8f_zero(), accH = v8f_zero();
  const __bf16* arow = hp + n * H_STRIDE;
  const __bf16* wcp = whc + (size_t)(n0 + n) * HID + 16 * hi;
  const __bf16* whp = whh + (size_t)(n0 + n) * HID + 16 * hi;
#pragma unroll 4
  for (int kk = 0; kk < 16; ++kk) {
    const int k0 = kk * 32;
    FragA a;
    a.h[0] = *(const v8bf*)(arow + k0 + 16 * hi);
    a.h[1] = *(const v8bf*)(arow + k0 + 16 * hi + 8);
    v16bf bc = *(const v16bf*)(wcp + k0);
    v16bf bh = *(const v16bf*)(whp + k0);
    accC = wmma_bf16(a.v, bc, accC);
    accH = wmma_bf16(a.v, bh, accH);
  }
#pragma unroll
  for (int v = 0; v < 8; ++v) {
    const int m = v + 8 * hi, col = n0 + n;
    float hpv = (float)hp[m * H_STRIDE + col];
    float c = sigmoidf_((float)xc[m * HID + col] + accC[v]);
    float g = tanhf((float)xh[m * HID + col] + accH[v]);
    hd[m * H_STRIDE + col] = (__bf16)(c * g + (1.0f - c) * hpv);
  }
  __syncthreads();
}

__device__ void output_node(const __bf16* h3, const __bf16* h6, const __bf16* h7,
                            __bf16* hout, __bf16* __restrict__ outs_t,
                            float* nsq, float* scl, int lane, int wave) {
  const int n = lane & 15, hi = lane >> 4;
  const int n0 = wave << 4;
  float o[8];
#pragma unroll
  for (int v = 0; v < 8; ++v) {
    const int idx = (v + 8 * hi) * H_STRIDE + n0 + n;
    o[v] = ((float)h3[idx] + (float)h6[idx] + (float)h7[idx]) * (1.0f / 3.0f);
  }
  if (threadIdx.x < 16) nsq[threadIdx.x] = 0.0f;
  __syncthreads();
#pragma unroll
  for (int v = 0; v < 8; ++v) atomicAdd(&nsq[v + 8 * hi], o[v] * o[v]);
  __syncthreads();
  if (threadIdx.x < 16) {
    float ns = nsq[threadIdx.x];
    scl[threadIdx.x] = (ns > 625.0f) ? 25.0f * rsqrtf(ns) : 1.0f;  // 25^2
  }
  __syncthreads();
#pragma unroll
  for (int v = 0; v < 8; ++v) {
    const int m = v + 8 * hi, col = n0 + n;
    float val = o[v] * scl[m];
    hout[m * H_STRIDE + col] = (__bf16)val;       // next h_prev
    outs_t[m * HID + col] = (__bf16)val;          // feed decoder
  }
  __syncthreads();
}

__global__ void __launch_bounds__(1024)
rnn_recurrence(const float* __restrict__ hidden,
               const __bf16* __restrict__ xpre,
               const __bf16* __restrict__ rec,
               __bf16* __restrict__ outs_bf) {
  extern __shared__ char smem[];
  __bf16* slot[5];
#pragma unroll
  for (int i = 0; i < 5; ++i) slot[i] = (__bf16*)(smem + (size_t)i * SLOT_BYTES);
  char* stage0 = smem + 5 * (size_t)SLOT_BYTES;
  char* stage1 = stage0 + STAGE_BYTES;
  float* nsq = (float*)(stage0 + 2 * (size_t)STAGE_BYTES);
  float* scl = nsq + 16;

  // kick off async stage of step 0 pre-activations, load initial hidden
  async_stage((const char*)xpre, stage0);
  for (int i = threadIdx.x; i < BATCH * HID; i += 1024) {
    int m = i >> 9, c = i & 511;
    slot[0][m * H_STRIDE + c] = (__bf16)hidden[i];
  }
  wait_async0();
  __syncthreads();

  const int lane = threadIdx.x & 31, wave = threadIdx.x >> 5;
  const int MS = HID * HID;
  // weight order: 0=w_hc, 1=w_hh, 2..8=edge_wc[0..6], 9..15=edge_wh[0..6]
  for (int t = 0; t < S_LEN; ++t) {
    char* cur = (t & 1) ? stage1 : stage0;
    char* nxt = (t & 1) ? stage0 : stage1;
    if (t + 1 < S_LEN)   // prefetch next step behind this step's 8K WMMAs
      async_stage((const char*)(xpre + (size_t)(t + 1) * 2 * BATCH * HID), nxt);
    const __bf16* xc = (const __bf16*)cur;
    const __bf16* xh = (const __bf16*)(cur + BATCH * HID * 2);
    // slot schedule (liveness): hprev->0, h0->1, h1->2, h2->3, h3->4,
    //                           h4->0, h5->1, h6->2, h7->3, out->0
    node0(slot[0], slot[1], rec + 0 * MS, rec + 1 * MS, xc, xh, lane, wave);
    dag_node(slot[1], slot[2], rec + (2 + 0) * MS, rec + (9 + 0) * MS, 0, lane, wave); // 0->1 tanh
    dag_node(slot[1], slot[3], rec + (2 + 1) * MS, rec + (9 + 1) * MS, 1, lane, wave); // 0->2 relu
    dag_node(slot[2], slot[4], rec + (2 + 2) * MS, rec + (9 + 2) * MS, 2, lane, wave); // 1->3 sigmoid
    dag_node(slot[2], slot[0], rec + (2 + 3) * MS, rec + (9 + 3) * MS, 0, lane, wave); // 1->4 tanh
    dag_node(slot[3], slot[1], rec + (2 + 4) * MS, rec + (9 + 4) * MS, 1, lane, wave); // 2->5 relu
    dag_node(slot[0], slot[2], rec + (2 + 5) * MS, rec + (9 + 5) * MS, 3, lane, wave); // 4->6 id
    dag_node(slot[1], slot[3], rec + (2 + 6) * MS, rec + (9 + 6) * MS, 0, lane, wave); // 5->7 tanh
    output_node(slot[4], slot[2], slot[3], slot[0],
                outs_bf + (size_t)t * BATCH * HID, nsq, scl, lane, wave);
    wait_async0();       // staged block for t+1 fully in LDS
    __syncthreads();
  }
}

// ---------------------------------------------------------------------------
// Kernel 3: decoder GEMM  [4096x512] @ [512x32000] + bias -> fp32 logits
// Each wave computes a 16x128 strip (8 WMMA tiles, A-fragment reused 8x).
// dec_w bf16 (32 MB) stays L2-resident; logits stores are non-temporal so
// the 524 MB writeback does not evict the weight working set.
// ---------------------------------------------------------------------------
__global__ void __launch_bounds__(256)
decoder_gemm(const __bf16* __restrict__ A, const __bf16* __restrict__ W,
             const float* __restrict__ bias, float* __restrict__ out) {
  const int wid = blockIdx.x * 8 + (threadIdx.x >> 5);  // 64000 wave-jobs
  const int lane = threadIdx.x & 31;
  const int n = lane & 15, hi = lane >> 4;
  const int mt = wid & 255;   // 256 M-tiles
  const int ng = wid >> 8;    // 250 N-groups of 128 cols
  const int m0 = mt << 4;
  const int jb = ng << 7;

  v8f acc[8];
#pragma unroll
  for (int u = 0; u < 8; ++u) acc[u] = v8f_zero();

  const __bf16* arow = A + (size_t)(m0 + n) * HID;
#pragma unroll 2
  for (int kk = 0; kk < 16; ++kk) {
    const int k0 = kk * 32;
    FragA a;
    a.h[0] = *(const v8bf*)(arow + k0 + 16 * hi);
    a.h[1] = *(const v8bf*)(arow + k0 + 16 * hi + 8);
#pragma unroll
    for (int u = 0; u < 8; ++u) {
      const __bf16* bp = W + (size_t)(jb + (u << 4) + n) * HID + k0 + 16 * hi;
      __builtin_prefetch(bp + 32, 0, 1);   // global_prefetch_b8 (next K chunk)
      v16bf b = *(const v16bf*)bp;
      acc[u] = wmma_bf16(a.v, b, acc[u]);
    }
  }
#pragma unroll
  for (int u = 0; u < 8; ++u) {
    const int j = jb + (u << 4) + n;
    const float bj = bias[j];
#pragma unroll
    for (int v = 0; v < 8; ++v) {
      const int m = m0 + v + 8 * hi;
      __builtin_nontemporal_store(acc[u][v] + bj, &out[(size_t)m * VOCAB + j]);
    }
  }
}

// ---------------------------------------------------------------------------
extern "C" void kernel_launch(void* const* d_in, const int* in_sizes, int n_in,
                              void* d_out, int out_size, void* d_ws, size_t ws_size,
                              hipStream_t stream) {
  (void)in_sizes; (void)n_in; (void)out_size; (void)ws_size;
  const int*   ids    = (const int*)  d_in[0];
  const float* hidden = (const float*)d_in[1];
  const float* emb    = (const float*)d_in[2];
  const float* w_xc   = (const float*)d_in[3];
  const float* b_xc   = (const float*)d_in[4];
  const float* w_xh   = (const float*)d_in[5];
  const float* b_xh   = (const float*)d_in[6];
  const float* w_hc   = (const float*)d_in[7];
  const float* w_hh   = (const float*)d_in[8];
  const float* e_wc   = (const float*)d_in[9];
  const float* e_wh   = (const float*)d_in[10];
  const float* dec_w  = (const float*)d_in[11];
  const float* dec_b  = (const float*)d_in[12];
  float* logits = (float*)d_out;

  char* ws = (char*)d_ws;
  size_t off = 0;
  __bf16* xpre    = (__bf16*)(ws + off); off += (size_t)S_LEN * 2 * BATCH * HID * 2;
  __bf16* rec_bf  = (__bf16*)(ws + off); off += (size_t)16 * HID * HID * 2;
  __bf16* dec_bf  = (__bf16*)(ws + off); off += (size_t)VOCAB * HID * 2;
  __bf16* outs_bf = (__bf16*)(ws + off); off += (size_t)S_LEN * BATCH * HID * 2;

  const int MS = HID * HID;
  f32_to_bf16<<<512,  256, 0, stream>>>(w_hc, rec_bf,          MS);
  f32_to_bf16<<<512,  256, 0, stream>>>(w_hh, rec_bf + MS,     MS);
  f32_to_bf16<<<2048, 256, 0, stream>>>(e_wc, rec_bf + 2 * MS, 7 * MS);
  f32_to_bf16<<<2048, 256, 0, stream>>>(e_wh, rec_bf + 9 * MS, 7 * MS);
  f32_to_bf16<<<4096, 256, 0, stream>>>(dec_w, dec_bf, VOCAB * HID);

  embed_xproj<<<dim3(S_LEN, HID / 16, 2), 32, 0, stream>>>(
      ids, emb, w_xc, b_xc, w_xh, b_xh, xpre);

  const size_t smem = 5 * (size_t)SLOT_BYTES + 2 * (size_t)STAGE_BYTES +
                      32 * sizeof(float);
  rnn_recurrence<<<1, 1024, smem, stream>>>(hidden, xpre, rec_bf, outs_bf);

  decoder_gemm<<<8000, 256, 0, stream>>>(outs_bf, dec_bf, dec_b, logits);
}
